// GridTransformerBlock_41120016892912
// MI455X (gfx1250) — compile-verified, matrix operand order checked by reading, not run on hardware
//
#include <hip/hip_runtime.h>
#include <hip/hip_bf16.h>

// ---------------------------------------------------------------------------
// GridTransformerBlock for MI455X (gfx1250, wave32, WMMA bf16 -> f32 accum)
//   B=8, C=256, H=W=128, G=16  -> 512 windows, N=256 tokens, C8=32, F=1024
// Token-major reformulation:  O = softmax((X Wq^T)(X Wk^T)^T) X Wv^T
//   with reassociation O = (P X) Wv^T so V is never materialized.
// All WMMA operand loads are contiguous 16B chunks:
//   A-fragments  <- [M][K] row-major storage
//   B-fragments  <- [N][K] storage (weights pre-swizzled to fragment order)
// Window staging uses the Tensor Data Mover when the builtin is available.
// ---------------------------------------------------------------------------

typedef __attribute__((ext_vector_type(16))) __bf16 v16bf;
typedef __attribute__((ext_vector_type(8)))  float  v8f;
typedef unsigned int u32x4 __attribute__((ext_vector_type(4)));
typedef int          i32x4 __attribute__((ext_vector_type(4)));
typedef int          i32x8 __attribute__((ext_vector_type(8)));

#define C_DIM   256
#define C8_DIM  32
#define N_TOK   256
#define F_DIM   1024
#define NWIN    512     // 8 * 8 * 8
#define NPIX    131072  // 8 * 128 * 128

__device__ __forceinline__ v8f wmma_bf16(v16bf a, v16bf b, v8f c) {
  return __builtin_amdgcn_wmma_f32_16x16x32_bf16(false, a, false, b, (short)0, c,
                                                 false, false);
}

// Fragment element e maps to k = k0 + half*8 + (e<8 ? e : 8+e): two 16B chunks
// at base and base+16 elements (ISA 7.12.2, 16-bit 16x32 operand layout).
__device__ __forceinline__ v16bf load_frag_gap(const __bf16* p) {
  v16bf r;
  *(uint4*)&r       = *(const uint4*)p;         // elements 0..7  <- p[0..7]
  *((uint4*)&r + 1) = *(const uint4*)(p + 16);  // elements 8..15 <- p[16..23]
  return r;
}

// A fragment (16x32): rows on lanes, from row-major [M][K] storage.
__device__ __forceinline__ v16bf load_frag_a(const __bf16* s, int ld, int row0,
                                             int k0, int lane) {
  int r = lane & 15, half = lane >> 4;
  return load_frag_gap(s + (size_t)(row0 + r) * ld + k0 + half * 8);
}

// B fragment (32x16): cols on lanes, from [N][K] storage (value(k,n)=s[n*ld+k]).
__device__ __forceinline__ v16bf load_frag_bT(const __bf16* s, int ld, int k0,
                                              int col0, int lane) {
  int n = lane & 15, half = lane >> 4;
  return load_frag_gap(s + (size_t)(col0 + n) * ld + k0 + half * 8);
}

// B fragment from pre-swizzled weights: 32B contiguous per lane, coalesced.
__device__ __forceinline__ v16bf load_frag_w(const __bf16* s, int ktiles,
                                             int kt, int nt, int lane) {
  const __bf16* p = s + (((size_t)(nt * ktiles + kt) * 32 + lane) << 4);
  v16bf r;
  *(uint4*)&r       = *(const uint4*)p;
  *((uint4*)&r + 1) = *(const uint4*)(p + 8);
  return r;
}

// Store 16x16 f32 accumulator tile (D layout: N on lanes, M on vgpr idx) as bf16.
__device__ __forceinline__ void store_tile_bf16(__bf16* d, int ld, int row0,
                                                int col0, v8f acc, int lane) {
  int n = lane & 15, half = lane >> 4;
#pragma unroll
  for (int r = 0; r < 8; ++r)
    d[(size_t)(row0 + r + 8 * half) * ld + col0 + n] = (__bf16)acc[r];
}

// ---------------------------------------------------------------------------
// Layout / conversion kernels
// ---------------------------------------------------------------------------

// Swizzle fp32 weight W[N][K] into bf16 fragment-linear order:
//   dst[((n>>4)*(K/32) + (k>>5))*512 + lane*16 + e]
__global__ void swz_w_kernel(const float* __restrict__ W,
                             __bf16* __restrict__ dst, int N, int K) {
  int i = blockIdx.x * 256 + threadIdx.x;
  if (i >= N * K) return;
  int n = i / K, k = i % K;
  int kl = k & 31, kt = k >> 5;
  int half, e;
  if (kl < 16) { half = kl >> 3; e = kl & 7; }
  else         { half = (kl - 16) >> 3; e = 8 + ((kl - 16) & 7); }
  int lane = (n & 15) + 16 * half;
  size_t d = ((size_t)((n >> 4) * (K >> 5) + kt) * 32 + lane) * 16 + e;
  dst[d] = (__bf16)W[i];
}

// NCHW x -> token-major bf16 [win][tok][chan], col-major bf16 [win][chan][tok],
// and token-major f32 (residual path).
__global__ void gather_x_kernel(const float* __restrict__ x,
                                __bf16* __restrict__ xt,
                                __bf16* __restrict__ xtt,
                                float* __restrict__ xf) {
  size_t i = (size_t)blockIdx.x * 256 + threadIdx.x;
  int c = (int)(i & 255);
  size_t p = i >> 8;
  int t = (int)(p & 255), wi = (int)(p >> 8);
  int b = wi >> 6, ih = (wi >> 3) & 7, iw = wi & 7;
  int h = ih * 16 + (t >> 4), w = iw * 16 + (t & 15);
  float v = x[(((size_t)b * C_DIM + c) * 128 + h) * 128 + w];
  xt[i] = (__bf16)v;
  xf[i] = v;
  xtt[((size_t)wi * C_DIM + c) * N_TOK + t] = (__bf16)v;
}

// ---------------------------------------------------------------------------
// Window attention: one block per window, 8 waves (256 threads).
// LDS: Xs col-major [chan][tok] 128K | Qs [tok][32] 16K | Ks [tok][32] 16K
//      Ps per-wave [16][256] 64K  => 229376 B (< 320 KB WGP LDS)
// ---------------------------------------------------------------------------
#define SM_X  0
#define SM_Q  131072
#define SM_K  (131072 + 16384)
#define SM_P  (131072 + 32768)
#define SM_ATTN_BYTES (131072 + 32768 + 65536)

__global__ void __launch_bounds__(256)
attn_kernel(const __bf16* __restrict__ Xtok, const __bf16* __restrict__ XtokT,
            const float* __restrict__ Xf32, const __bf16* __restrict__ Wqs,
            const __bf16* __restrict__ Wks, const __bf16* __restrict__ Wvs,
            const float* __restrict__ gamma_p, float* __restrict__ xa_f32,
            __bf16* __restrict__ xa_bf16) {
  extern __shared__ char smem[];
  __bf16* Xs = (__bf16*)(smem + SM_X);   // col-major: Xs[c*256 + t]
  __bf16* Qs = (__bf16*)(smem + SM_Q);
  __bf16* Ks = (__bf16*)(smem + SM_K);
  __bf16* Ps = (__bf16*)(smem + SM_P);

  const int wi = blockIdx.x;
  const int tid = threadIdx.x;
  const int lane = tid & 31;
  const int wid = tid >> 5;
  const float gamma = gamma_p[0];

  const __bf16* XtokW = Xtok + (size_t)wi * N_TOK * C_DIM;   // row-major window
  const __bf16* gsrc  = XtokT + (size_t)wi * N_TOK * C_DIM;  // col-major window

  // ---- Stage col-major window X into LDS -------------------------------
#if defined(__gfx1250__) && __has_builtin(__builtin_amdgcn_tensor_load_to_lds)
  // Tensor Data Mover: 2-D tile 256x256, 2-byte elements, dense strides.
  if (tid == 0) {
    unsigned long long ga = (unsigned long long)(size_t)gsrc;
    u32x4 g0;
    g0[0] = 1u;                                   // count=1, user descriptor
    g0[1] = 0u;                                   // lds_addr = 0 (Xs base)
    g0[2] = (unsigned)(ga & 0xffffffffu);         // global_addr[31:0]
    g0[3] = (unsigned)((ga >> 32) & 0x01ffffffu)  // global_addr[56:32]
            | (2u << 30);                         // type = 2 ("image")
    i32x8 g1;
    g1[0] = 0x00010000;  // wg_mask=0, data_size=1 (2 bytes)
    g1[1] = 0x01000000;  // tensor_dim0 = 256 (low 16 bits in [31:16])
    g1[2] = 0x01000000;  // tensor_dim0 hi=0, tensor_dim1 = 256 (low 16)
    g1[3] = 0x01000000;  // tensor_dim1 hi=0, tile_dim0 = 256
    g1[4] = 0x00000100;  // tile_dim1 = 256, tile_dim2 = 0
    g1[5] = 0x00000100;  // tensor_dim0_stride = 256
    g1[6] = 0;
    g1[7] = 0;
    i32x4 gz = {0, 0, 0, 0};
#if __clang_major__ >= 23
    i32x8 gz8 = {0, 0, 0, 0, 0, 0, 0, 0};
    __builtin_amdgcn_tensor_load_to_lds(g0, g1, gz, gz, gz8, 0);
#else
    __builtin_amdgcn_tensor_load_to_lds(g0, g1, gz, gz, 0);
#endif
    __builtin_amdgcn_s_wait_tensorcnt(0);
  }
#else
  {
    const uint4* s4 = (const uint4*)gsrc;
    uint4* d4 = (uint4*)Xs;
    for (int i = tid; i < (N_TOK * C_DIM) / 8; i += 256) d4[i] = s4[i];
  }
#endif
  __syncthreads();

  // ---- Cooperative Q = X*Wq^T, K = X*Wk^T ([256,32]) into LDS ----------
  // A-fragments straight from the row-major global window (contiguous 16B).
  for (int tt = wid * 2; tt < wid * 2 + 2; ++tt) {
    for (int nt = 0; nt < 2; ++nt) {
      v8f accq = {};
      v8f acck = {};
#pragma unroll
      for (int ks = 0; ks < 8; ++ks) {
        v16bf a  = load_frag_a(XtokW, C_DIM, tt * 16, ks * 32, lane);
        v16bf bq = load_frag_w(Wqs, 8, ks, nt, lane);
        v16bf bk = load_frag_w(Wks, 8, ks, nt, lane);
        accq = wmma_bf16(a, bq, accq);
        acck = wmma_bf16(a, bk, acck);
      }
      store_tile_bf16(Qs, C8_DIM, tt * 16, nt * 16, accq, lane);
      store_tile_bf16(Ks, C8_DIM, tt * 16, nt * 16, acck, lane);
    }
  }
  __syncthreads();

  __bf16* Pw = Ps + (size_t)wid * 16 * N_TOK;  // per-wave P / T scratch
  const int b = wi >> 6, ih = (wi >> 3) & 7, iw = wi & 7;

  for (int rt = wid; rt < 16; rt += 8) {
    const int row0 = rt * 16;
    v8f acc[16];

    // S = Q * K^T (K=32 covers the whole C/8 dim; Ks is [n=tok][k=o])
    {
      v16bf aq = load_frag_a(Qs, C8_DIM, row0, 0, lane);
#pragma unroll
      for (int nt = 0; nt < 16; ++nt) {
        v8f z = {};
        v16bf bk = load_frag_bT(Ks, C8_DIM, 0, nt * 16, lane);
        acc[nt] = wmma_bf16(aq, bk, z);
      }
    }

    // Row softmax over 256 keys (row r+8*half lives on 16 lanes of a half).
    float rinv[8];
#pragma unroll
    for (int r = 0; r < 8; ++r) {
      float m = acc[0][r];
#pragma unroll
      for (int nt = 1; nt < 16; ++nt) m = fmaxf(m, acc[nt][r]);
      for (int off = 1; off < 16; off <<= 1) m = fmaxf(m, __shfl_xor(m, off, 32));
      float s = 0.f;
#pragma unroll
      for (int nt = 0; nt < 16; ++nt) {
        float e = __expf(acc[nt][r] - m);
        acc[nt][r] = e;
        s += e;
      }
      for (int off = 1; off < 16; off <<= 1) s += __shfl_xor(s, off, 32);
      rinv[r] = 1.0f / s;
    }
    for (int nt = 0; nt < 16; ++nt) {
#pragma unroll
      for (int r = 0; r < 8; ++r) acc[nt][r] *= rinv[r];
      store_tile_bf16(Pw, N_TOK, 0, nt * 16, acc[nt], lane);
    }

    // T = P * X  [16,256]; B from col-major Xs ([n=chan][k=tok], contiguous)
    for (int nt = 0; nt < 16; ++nt) { v8f z = {}; acc[nt] = z; }
    for (int ks = 0; ks < 8; ++ks) {
      v16bf ap = load_frag_a(Pw, N_TOK, 0, ks * 32, lane);
#pragma unroll
      for (int nt = 0; nt < 16; ++nt) {
        v16bf bx = load_frag_bT(Xs, N_TOK, ks * 32, nt * 16, lane);
        acc[nt] = wmma_bf16(ap, bx, acc[nt]);
      }
    }
    for (int nt = 0; nt < 16; ++nt)
      store_tile_bf16(Pw, N_TOK, 0, nt * 16, acc[nt], lane);

    // O = T * Wv^T  [16,256]; B from swizzled Wv (coalesced)
    for (int nt = 0; nt < 16; ++nt) { v8f z = {}; acc[nt] = z; }
    for (int ks = 0; ks < 8; ++ks) {
      v16bf at = load_frag_a(Pw, N_TOK, 0, ks * 32, lane);
#pragma unroll
      for (int nt = 0; nt < 16; ++nt) {
        v16bf bv = load_frag_w(Wvs, 8, ks, nt, lane);
        acc[nt] = wmma_bf16(at, bv, acc[nt]);
      }
    }

    // aw = gamma*O + x (f32 residual, coalesced token-major); dual output.
    for (int nt = 0; nt < 16; ++nt) {
      const int c = nt * 16 + (lane & 15);
#pragma unroll
      for (int r = 0; r < 8; ++r) {
        const int t = row0 + r + 8 * (lane >> 4);
        const size_t p = (size_t)wi * N_TOK + t;
        const float val = gamma * acc[nt][r] + Xf32[p * C_DIM + c];
        xa_f32[p * C_DIM + c] = val;
        xa_bf16[p * C_DIM + c] = (__bf16)val;
      }
    }
  }
}

// ---------------------------------------------------------------------------
// Fused FFN + 2x channel-LayerNorm + residuals; 32 tokens per block.
// ---------------------------------------------------------------------------
#define FSM_XA 0
#define FSM_H1 16384
#define FSM_Y  (16384 + 65536)
#define SM_FFN_BYTES (16384 + 65536 + 32768)

__global__ void __launch_bounds__(256)
ffn_kernel(const float* __restrict__ xa_f32, const __bf16* __restrict__ xa_bf16,
           const __bf16* __restrict__ W1s, const float* __restrict__ b1,
           const __bf16* __restrict__ W2s, const float* __restrict__ b2,
           const float* __restrict__ ln1w, const float* __restrict__ ln1b,
           const float* __restrict__ ln2w, const float* __restrict__ ln2b,
           float* __restrict__ out) {
  extern __shared__ char smem[];
  __bf16* XaS = (__bf16*)(smem + FSM_XA);   // [32][256] bf16
  __bf16* H1S = (__bf16*)(smem + FSM_H1);   // [32][1024] bf16
  float*  YS  = (float*)(smem + FSM_Y);     // [32][256] f32

  const int tid = threadIdx.x, lane = tid & 31, wid = tid >> 5;
  const size_t p0 = (size_t)blockIdx.x * 32;

  {
    const uint4* s4 = (const uint4*)(xa_bf16 + p0 * C_DIM);
    uint4* d4 = (uint4*)XaS;
    for (int i = tid; i < (32 * C_DIM) / 8; i += 256) d4[i] = s4[i];
  }
  __syncthreads();

  // GEMM1: H1 = gelu(Xa * W1^T + b1)  [32,1024]; W1 swizzled, ktiles=8
  for (int it = 0; it < 16; ++it) {
    const int idx = wid + it * 8;  // 0..127 -> (mt, nt)
    const int mt = idx & 1, nt = idx >> 1;
    v8f acc = {};
#pragma unroll
    for (int ks = 0; ks < 8; ++ks) {
      v16bf a  = load_frag_a(XaS, C_DIM, mt * 16, ks * 32, lane);
      v16bf bw = load_frag_w(W1s, 8, ks, nt, lane);
      acc = wmma_bf16(a, bw, acc);
    }
    const float bias = b1[nt * 16 + (lane & 15)];
#pragma unroll
    for (int r = 0; r < 8; ++r) {
      float v = acc[r] + bias;
      acc[r] = 0.5f * v * (1.0f + erff(v * 0.70710678118654752f));  // exact gelu
    }
    store_tile_bf16(H1S, F_DIM, mt * 16, nt * 16, acc, lane);
  }
  __syncthreads();

  // GEMM2: Y = H1 * W2^T + b2  [32,256]; W2 swizzled, ktiles=32
  for (int it = 0; it < 4; ++it) {
    const int idx = wid + it * 8;  // 0..31
    const int mt = idx & 1, nt = idx >> 1;
    v8f acc = {};
    for (int ks = 0; ks < 32; ++ks) {
      v16bf a  = load_frag_a(H1S, F_DIM, mt * 16, ks * 32, lane);
      v16bf bw = load_frag_w(W2s, 32, ks, nt, lane);
      acc = wmma_bf16(a, bw, acc);
    }
    const int c = nt * 16 + (lane & 15);
    const float bias = b2[c];
#pragma unroll
    for (int r = 0; r < 8; ++r)
      YS[(size_t)(mt * 16 + r + 8 * (lane >> 4)) * C_DIM + c] = acc[r] + bias;
  }
  __syncthreads();

  // x2 = xa + LN1(y); out = x2 + LN2(x2). 8 threads per token row.
  const int row = tid >> 3;
  const int sub = tid & 7;
  float y[32], x2[32];
  float s1 = 0.f, s2 = 0.f;
#pragma unroll
  for (int j = 0; j < 32; ++j) {
    float v = YS[(size_t)row * C_DIM + sub * 32 + j];
    y[j] = v; s1 += v; s2 += v * v;
  }
  for (int off = 1; off < 8; off <<= 1) {
    s1 += __shfl_xor(s1, off, 32);
    s2 += __shfl_xor(s2, off, 32);
  }
  const float mean1 = s1 * (1.0f / 256.0f);
  const float rstd1 = rsqrtf(s2 * (1.0f / 256.0f) - mean1 * mean1 + 1e-5f);

  const size_t p = p0 + row;
  float t1 = 0.f, t2 = 0.f;
#pragma unroll
  for (int j = 0; j < 32; ++j) {
    const int c = sub * 32 + j;
    float v = xa_f32[p * C_DIM + c] + (y[j] - mean1) * rstd1 * ln1w[c] + ln1b[c];
    x2[j] = v; t1 += v; t2 += v * v;
  }
  for (int off = 1; off < 8; off <<= 1) {
    t1 += __shfl_xor(t1, off, 32);
    t2 += __shfl_xor(t2, off, 32);
  }
  const float mean2 = t1 * (1.0f / 256.0f);
  const float rstd2 = rsqrtf(t2 * (1.0f / 256.0f) - mean2 * mean2 + 1e-5f);

  // token -> NCHW scatter
  const int wi = (int)(p >> 8), t = (int)(p & 255);
  const int b = wi >> 6, ih = (wi >> 3) & 7, iw = wi & 7;
  const int h = ih * 16 + (t >> 4), w = iw * 16 + (t & 15);
#pragma unroll
  for (int j = 0; j < 32; ++j) {
    const int c = sub * 32 + j;
    out[(((size_t)b * C_DIM + c) * 128 + h) * 128 + w] =
        x2[j] + (x2[j] - mean2) * rstd2 * ln2w[c] + ln2b[c];
  }
}

// ---------------------------------------------------------------------------
// Host side: ws layout + launches (all on `stream`, graph-capture safe).
// ---------------------------------------------------------------------------
#define OFF_WQ   ((size_t)0)                                // 16 KiB
#define OFF_WK   ((size_t)16384)                            // 16 KiB
#define OFF_WV   ((size_t)32768)                            // 128 KiB
#define OFF_W1   ((size_t)163840)                           // 512 KiB
#define OFF_W2   ((size_t)688128)                           // 512 KiB
#define OFF_XT   ((size_t)1212416)                          // 64 MiB bf16 row
#define OFF_XTT  (OFF_XT  + (size_t)NPIX * C_DIM * 2)       // 64 MiB bf16 col
#define OFF_XF   (OFF_XTT + (size_t)NPIX * C_DIM * 2)       // 128 MiB f32
#define OFF_XA32 (OFF_XF  + (size_t)NPIX * C_DIM * 4)       // 128 MiB f32
#define OFF_XA16 (OFF_XA32 + (size_t)NPIX * C_DIM * 4)      // 64 MiB bf16

extern "C" void kernel_launch(void* const* d_in, const int* in_sizes, int n_in,
                              void* d_out, int out_size, void* d_ws,
                              size_t ws_size, hipStream_t stream) {
  const float* x     = (const float*)d_in[0];
  const float* Wq    = (const float*)d_in[1];
  const float* Wk    = (const float*)d_in[2];
  const float* Wv    = (const float*)d_in[3];
  const float* gamma = (const float*)d_in[4];
  const float* W1    = (const float*)d_in[5];
  const float* b1    = (const float*)d_in[6];
  const float* W2    = (const float*)d_in[7];
  const float* b2    = (const float*)d_in[8];
  const float* ln1w  = (const float*)d_in[9];
  const float* ln1b  = (const float*)d_in[10];
  const float* ln2w  = (const float*)d_in[11];
  const float* ln2b  = (const float*)d_in[12];
  float* out = (float*)d_out;

  char* ws = (char*)d_ws;
  __bf16* Wqs  = (__bf16*)(ws + OFF_WQ);
  __bf16* Wks  = (__bf16*)(ws + OFF_WK);
  __bf16* Wvs  = (__bf16*)(ws + OFF_WV);
  __bf16* W1s  = (__bf16*)(ws + OFF_W1);
  __bf16* W2s  = (__bf16*)(ws + OFF_W2);
  __bf16* Xtok = (__bf16*)(ws + OFF_XT);
  __bf16* XtokT= (__bf16*)(ws + OFF_XTT);
  float*  Xf32 = (float*)(ws + OFF_XF);
  float*  xa32 = (float*)(ws + OFF_XA32);
  __bf16* xa16 = (__bf16*)(ws + OFF_XA16);

  // Stage 0: weight swizzle (fragment-linear bf16) + x layout conversion
  swz_w_kernel<<<(C8_DIM * C_DIM + 255) / 256, 256, 0, stream>>>(Wq, Wqs, C8_DIM, C_DIM);
  swz_w_kernel<<<(C8_DIM * C_DIM + 255) / 256, 256, 0, stream>>>(Wk, Wks, C8_DIM, C_DIM);
  swz_w_kernel<<<(C_DIM * C_DIM + 255) / 256, 256, 0, stream>>>(Wv, Wvs, C_DIM, C_DIM);
  swz_w_kernel<<<(F_DIM * C_DIM + 255) / 256, 256, 0, stream>>>(W1, W1s, F_DIM, C_DIM);
  swz_w_kernel<<<(C_DIM * F_DIM + 255) / 256, 256, 0, stream>>>(W2, W2s, C_DIM, F_DIM);
  gather_x_kernel<<<(int)(((size_t)NPIX * C_DIM) / 256), 256, 0, stream>>>(
      x, Xtok, XtokT, Xf32);

  // Stage 1: windowed attention (one block per 16x16 window)
  attn_kernel<<<NWIN, 256, SM_ATTN_BYTES, stream>>>(Xtok, XtokT, Xf32, Wqs, Wks,
                                                    Wvs, gamma, xa32, xa16);

  // Stage 2: fused FFN + LayerNorms + residuals (32 tokens per block)
  ffn_kernel<<<NPIX / 32, 256, SM_FFN_BYTES, stream>>>(xa32, xa16, W1s, b1, W2s,
                                                       b2, ln1w, ln1b, ln2w,
                                                       ln2b, out);
}
